// ContrastiveLoss_6932077215890
// MI455X (gfx1250) — matrix operand order repaired and checked by reading, not compiled, not run on hardware
//
#include <hip/hip_runtime.h>
#include <hip/hip_bf16.h>

// ContrastiveLoss on MI455X (gfx1250).
// Strategy: split-bf16 (hi+lo) emulated-fp32 GEMM via v_wmma_f32_16x16x32_bf16,
// fused with exp-sum row/col reductions (no max shift needed: |logit| <= 14.3).
// bf16 hi/lo data (64MB total) is L2-resident (192MB L2), so fragments are
// loaded straight from global in contiguous-k fragment order (k = 16*h + s).
//
// Workspace layout (needs ~64.1 MB):
//   img_hi, img_lo, txt_hi, txt_lo : bf16 [8192*1024] each
//   rowsum, colsum, diag           : f32  [8192] each

#define NROWS 8192
#define DIMK  1024
#define INV_T 14.2857142857142857f   // 1/0.07

typedef __attribute__((ext_vector_type(16))) __bf16 v16bf;
typedef __attribute__((ext_vector_type(8)))  __bf16 bf16x8;
typedef __attribute__((ext_vector_type(8)))  float  v8f;

union AFrag { v16bf v; bf16x8 h[2]; };

// ---------------------------------------------------------------- zero sums
__global__ void cl_zero_kernel(float* __restrict__ rowsum, float* __restrict__ colsum) {
    int i = blockIdx.x * blockDim.x + threadIdx.x;
    if (i < NROWS) { rowsum[i] = 0.0f; colsum[i] = 0.0f; }
}

// ------------------------------------------------- normalize + bf16 hi/lo split
__global__ void cl_norm_split_kernel(const float* __restrict__ img,
                                     const float* __restrict__ txt,
                                     __bf16* __restrict__ img_hi, __bf16* __restrict__ img_lo,
                                     __bf16* __restrict__ txt_hi, __bf16* __restrict__ txt_lo) {
    __shared__ float sdata[256];
    const int row = blockIdx.x;
    const int sel = blockIdx.y;                 // 0 = image, 1 = text
    const float* __restrict__ src = sel ? txt : img;
    __bf16* __restrict__ hi = sel ? txt_hi : img_hi;
    __bf16* __restrict__ lo = sel ? txt_lo : img_lo;
    const int tid = threadIdx.x;

    float x[4]; float ss = 0.0f;
#pragma unroll
    for (int k = 0; k < 4; ++k) {
        x[k] = src[(size_t)row * DIMK + tid + k * 256];
        ss += x[k] * x[k];
    }
    sdata[tid] = ss; __syncthreads();
    for (int s = 128; s > 0; s >>= 1) {
        if (tid < s) sdata[tid] += sdata[tid + s];
        __syncthreads();
    }
    const float inv = 1.0f / fmaxf(sqrtf(sdata[0]), 1e-8f);
#pragma unroll
    for (int k = 0; k < 4; ++k) {
        float v = x[k] * inv;
        __bf16 h = (__bf16)v;          // RTNE high part
        float  r = v - (float)h;       // residual captures 8 more mantissa bits
        hi[(size_t)row * DIMK + tid + k * 256] = h;
        lo[(size_t)row * DIMK + tid + k * 256] = (__bf16)r;
    }
}

// -------------------------------------------------------- exact fp32 diagonal
__global__ void cl_diag_kernel(const float* __restrict__ img,
                               const float* __restrict__ txt,
                               float* __restrict__ diag) {
    __shared__ float sii[256], stt[256], sit[256];
    const int row = blockIdx.x;
    const int tid = threadIdx.x;
    float aii = 0.f, att = 0.f, ait = 0.f;
#pragma unroll
    for (int k = 0; k < 4; ++k) {
        float a = img[(size_t)row * DIMK + tid + k * 256];
        float b = txt[(size_t)row * DIMK + tid + k * 256];
        aii += a * a; att += b * b; ait += a * b;
    }
    sii[tid] = aii; stt[tid] = att; sit[tid] = ait; __syncthreads();
    for (int s = 128; s > 0; s >>= 1) {
        if (tid < s) { sii[tid] += sii[tid+s]; stt[tid] += stt[tid+s]; sit[tid] += sit[tid+s]; }
        __syncthreads();
    }
    if (tid == 0) {
        float na = fmaxf(sqrtf(sii[0]), 1e-8f);
        float nb = fmaxf(sqrtf(stt[0]), 1e-8f);
        diag[row] = (sit[0] / (na * nb)) * INV_T;
    }
}

// ------------------------------------- fused split-bf16 GEMM + exp row/col sums
// Block = 256 threads = 8 waves; WG tile 128 (rows) x 256 (cols).
// Wave tile 64x64 = 4x4 WMMA 16x16 tiles; K-step 32; 48 WMMAs per K-step.
__global__ __launch_bounds__(256)
void cl_gemm_lse_kernel(const __bf16* __restrict__ img_hi, const __bf16* __restrict__ img_lo,
                        const __bf16* __restrict__ txt_hi, const __bf16* __restrict__ txt_lo,
                        float* __restrict__ rowsum, float* __restrict__ colsum) {
    const unsigned lane = threadIdx.x & 31u;
    const unsigned wave = threadIdx.x >> 5;
    const unsigned wr = wave & 1u;
    const unsigned wc = wave >> 1;
    const int row0 = blockIdx.y * 128 + wr * 64;
    const int col0 = blockIdx.x * 256 + wc * 64;
    const unsigned g   = lane >> 4;     // half-wave group (K-group select)
    const unsigned l15 = lane & 15u;

    v8f acc[4][4];
#pragma unroll
    for (int mt = 0; mt < 4; ++mt)
#pragma unroll
        for (int nt = 0; nt < 4; ++nt)
#pragma unroll
            for (int r = 0; r < 8; ++r) acc[mt][nt][r] = 0.0f;

    for (int kb = 0; kb < DIMK; kb += 32) {
        // A fragments: 16x32 bf16, lane l row M=l&15 holds K = 16*(l>>4) + {0..15}
        AFrag ah[4], al[4];
#pragma unroll
        for (int mt = 0; mt < 4; ++mt) {
            size_t off = (size_t)(row0 + mt * 16 + l15) * DIMK + kb + 16 * g;
            const __bf16* p = img_hi + off;
            const __bf16* q = img_lo + off;
            ah[mt].h[0] = *(const bf16x8*)(p);
            ah[mt].h[1] = *(const bf16x8*)(p + 8);
            al[mt].h[0] = *(const bf16x8*)(q);
            al[mt].h[1] = *(const bf16x8*)(q + 8);
        }
        // B fragments: 32x16 bf16, lane l col N=l&15 holds K = 16*(l>>4) + {0..15}
        v16bf bh[4], bl[4];
#pragma unroll
        for (int nt = 0; nt < 4; ++nt) {
            size_t off = (size_t)(col0 + nt * 16 + l15) * DIMK + kb + 16 * g;
            bh[nt] = *(const v16bf*)(txt_hi + off);
            bl[nt] = *(const v16bf*)(txt_lo + off);
        }
        // Split-bf16 products: hi*hi + lo*hi + hi*lo (lo*lo ~2^-16 rel, dropped)
#pragma unroll
        for (int mt = 0; mt < 4; ++mt)
#pragma unroll
            for (int nt = 0; nt < 4; ++nt) {
                acc[mt][nt] = __builtin_amdgcn_wmma_f32_16x16x32_bf16(
                    false, ah[mt].v, false, bh[nt], (short)0, acc[mt][nt], false, false);
                acc[mt][nt] = __builtin_amdgcn_wmma_f32_16x16x32_bf16(
                    false, al[mt].v, false, bh[nt], (short)0, acc[mt][nt], false, false);
                acc[mt][nt] = __builtin_amdgcn_wmma_f32_16x16x32_bf16(
                    false, ah[mt].v, false, bl[nt], (short)0, acc[mt][nt], false, false);
            }
    }

    // Epilogue: e = exp(logit); reduce rows (across lanes) and cols (across regs).
    // C/D layout: VGPR r, lane l -> row = r + 8*(l>>4), col = l&15.
    float colp[4] = {0.f, 0.f, 0.f, 0.f};
#pragma unroll
    for (int mt = 0; mt < 4; ++mt) {
        float rowp[8];
#pragma unroll
        for (int r = 0; r < 8; ++r) rowp[r] = 0.0f;
#pragma unroll
        for (int nt = 0; nt < 4; ++nt)
#pragma unroll
            for (int r = 0; r < 8; ++r) {
                float e = __expf(acc[mt][nt][r] * INV_T);
                rowp[r] += e;        // partial over this lane's 4 columns
                colp[nt] += e;       // partial over this lane's 32 rows (half set)
            }
#pragma unroll
        for (int r = 0; r < 8; ++r) {
            float v = rowp[r];
            v += __shfl_xor(v, 1);
            v += __shfl_xor(v, 2);
            v += __shfl_xor(v, 4);
            v += __shfl_xor(v, 8);   // full 64-col row sum within each 16-lane group
            if (l15 == 0)
                unsafeAtomicAdd(&rowsum[row0 + mt * 16 + r + 8 * g], v);
        }
    }
#pragma unroll
    for (int nt = 0; nt < 4; ++nt) {
        float v = colp[nt];
        v += __shfl_xor(v, 16);      // combine the two row-half sets -> 64-row col sum
        if (lane < 16)
            unsafeAtomicAdd(&colsum[col0 + nt * 16 + l15], v);
    }
}

// ------------------------------------------------------------- final reduction
__global__ void cl_loss_kernel(const float* __restrict__ rowsum,
                               const float* __restrict__ colsum,
                               const float* __restrict__ diag,
                               float* __restrict__ out) {
    __shared__ float sdata[1024];
    const int tid = threadIdx.x;
    float p = 0.0f;
#pragma unroll
    for (int k = 0; k < 8; ++k) {
        int i = tid + k * 1024;
        p += 0.5f * (logf(rowsum[i]) + logf(colsum[i])) - diag[i];
    }
    sdata[tid] = p; __syncthreads();
    for (int s = 512; s > 0; s >>= 1) {
        if (tid < s) sdata[tid] += sdata[tid + s];
        __syncthreads();
    }
    if (tid == 0) out[0] = sdata[0] / (float)NROWS;
}

extern "C" void kernel_launch(void* const* d_in, const int* in_sizes, int n_in,
                              void* d_out, int out_size, void* d_ws, size_t ws_size,
                              hipStream_t stream) {
    const float* img = (const float*)d_in[0];
    const float* txt = (const float*)d_in[1];
    float* out = (float*)d_out;

    const size_t nd = (size_t)NROWS * DIMK;
    __bf16* img_hi = (__bf16*)d_ws;
    __bf16* img_lo = img_hi + nd;
    __bf16* txt_hi = img_lo + nd;
    __bf16* txt_lo = txt_hi + nd;
    float*  rowsum = (float*)(txt_lo + nd);
    float*  colsum = rowsum + NROWS;
    float*  diag   = colsum + NROWS;

    cl_zero_kernel<<<(NROWS + 255) / 256, 256, 0, stream>>>(rowsum, colsum);
    cl_norm_split_kernel<<<dim3(NROWS, 2), 256, 0, stream>>>(img, txt,
                                                             img_hi, img_lo, txt_hi, txt_lo);
    cl_diag_kernel<<<NROWS, 256, 0, stream>>>(img, txt, diag);
    cl_gemm_lse_kernel<<<dim3(NROWS / 256, NROWS / 128), 256, 0, stream>>>(
        img_hi, img_lo, txt_hi, txt_lo, rowsum, colsum);
    cl_loss_kernel<<<1, 1024, 0, stream>>>(rowsum, colsum, diag, out);
}
